// TransformerBlock3D_75539884802995
// MI455X (gfx1250) — matrix operand order repaired and checked by reading, not compiled, not run on hardware
//
#include <hip/hip_runtime.h>
#include <hip/hip_bf16.h>

// ---------------------------------------------------------------------------
// Types for CDNA5 WMMA (wave32)
// ---------------------------------------------------------------------------
typedef __attribute__((ext_vector_type(16))) __bf16          v16bf;
typedef __attribute__((ext_vector_type(8)))  float           v8f;
typedef __attribute__((ext_vector_type(16))) unsigned short  v16u;
typedef __attribute__((ext_vector_type(8)))  unsigned short  u16x8;

// fp32 -> bf16 (round to nearest even)
__device__ __forceinline__ unsigned short f2bf(float f) {
    unsigned int u = __float_as_uint(f);
    u += 0x7FFFu + ((u >> 16) & 1u);
    return (unsigned short)(u >> 16);
}

// ---------------------------------------------------------------------------
// CDNA5 async global->LDS copy (16B per lane), tracked by ASYNCcnt.
// vdst = LDS byte address (low 32 bits of flat shared pointer), vaddr = global.
// ---------------------------------------------------------------------------
__device__ __forceinline__ void async_copy_b128(void* lds_dst, const void* gsrc) {
    unsigned lds = (unsigned)(unsigned long long)lds_dst;
    asm volatile("global_load_async_to_lds_b128 %0, %1, off"
                 :: "v"(lds), "v"(gsrc)
                 : "memory");
}
__device__ __forceinline__ void wait_async0() {
    asm volatile("s_wait_asynccnt 0" ::: "memory");
}

// A fragment (16x32 bf16) from row-major LDS [.. , ld] at rows m, cols 0..31.
// Layout: lanes 0-15 hold M=lane, K=0..15; lanes 16-31 K=16..31.
__device__ __forceinline__ v16bf frag_a_lds(const unsigned short* Asrc, int m, int k0,
                                            int half, int ld) {
    const u16x8* p0 = (const u16x8*)(Asrc + m * ld + k0 + half * 16);
    const u16x8* p1 = (const u16x8*)(Asrc + m * ld + k0 + half * 16 + 8);
    u16x8 lo = *p0, hi = *p1;
    v16u r;
#pragma unroll
    for (int i = 0; i < 8; ++i) { r[i] = lo[i]; r[i + 8] = hi[i]; }
    return __builtin_bit_cast(v16bf, r);
}

// B fragment (32x16 bf16) from transposed LDS BsT[n][k] (stride ld).
// Layout: lanes 0-15 hold K=0..15 of column N=lane; lanes 16-31 hold K=16..31.
__device__ __forceinline__ v16bf frag_b_ldsT(const unsigned short* BsT, int n,
                                             int half, int ld) {
    const u16x8* p0 = (const u16x8*)(BsT + n * ld + half * 16);
    const u16x8* p1 = (const u16x8*)(BsT + n * ld + half * 16 + 8);
    u16x8 lo = *p0, hi = *p1;
    v16u r;
#pragma unroll
    for (int i = 0; i < 8; ++i) { r[i] = lo[i]; r[i + 8] = hi[i]; }
    return __builtin_bit_cast(v16bf, r);
}

__device__ __forceinline__ v8f wmma_bf16(v16bf a, v16bf b, v8f c) {
    return __builtin_amdgcn_wmma_f32_16x16x32_bf16(false, a, false, b,
                                                   (short)0, c, false, false);
}

// ---------------------------------------------------------------------------
// Generic bf16 WMMA GEMM:  out = epilogue(A[M,K] * B[K,N])
//   block tile 128x64, 8 waves of 32x32 (2x2 WMMA tiles), K step 32
//   A tile staged via async global->LDS (ASYNCcnt); B staged + transposed.
//   EPI 0: store bf16            (no bias)
//   EPI 1: +bias, exact GELU ->  store bf16
//   EPI 2: +bias, +residual  ->  store fp32
// M % 128 == 0, N % 64 == 0, K % 32 == 0 (true for all calls here).
// ---------------------------------------------------------------------------
#define GBM 128
#define GBN 64
#define GBK 32
#define GLD 40   // padded LDS stride (keeps 16B alignment, spreads banks)

template <int EPI>
__global__ __launch_bounds__(256)
void gemm_bf16_wmma(const unsigned short* __restrict__ A,
                    const unsigned short* __restrict__ B,
                    const float* __restrict__ bias,
                    const float* __restrict__ resid,
                    float* __restrict__ outF,
                    unsigned short* __restrict__ outH,
                    int M, int N, int K, int lda, int ldb, int ldc,
                    long strideA, long strideB, long strideC) {
    __shared__ unsigned short As[GBM * GLD];
    __shared__ unsigned short Bs[GBN * GLD];

    const int tid  = threadIdx.x;
    const int lane = tid & 31;
    const int w    = tid >> 5;
    const int half = lane >> 4;
    const int l16  = lane & 15;

    const int m0 = blockIdx.y * GBM;
    const int n0 = blockIdx.x * GBN;
    const int bz = blockIdx.z;
    A += (long)bz * strideA;
    B += (long)bz * strideB;
    const long cbase = (long)bz * strideC;

    const int wm0 = (w >> 1) * 32;  // wave M offset in block tile
    const int wn0 = (w & 1) * 32;   // wave N offset

    const v8f vzero = {0.f, 0.f, 0.f, 0.f, 0.f, 0.f, 0.f, 0.f};
    v8f acc[2][2];
#pragma unroll
    for (int i = 0; i < 2; ++i)
#pragma unroll
        for (int j = 0; j < 2; ++j) acc[i][j] = vzero;

    for (int kb = 0; kb < K; kb += GBK) {
        // ---- stage A tile 128x32 via async copy: 512 x 16B chunks ----
#pragma unroll
        for (int i = 0; i < 2; ++i) {
            int c   = tid + 256 * i;       // chunk id, 0..511
            int r   = c >> 2;              // row 0..127
            int c16 = (c & 3) * 8;         // 8 bf16 = 16B
            async_copy_b128(As + r * GLD + c16,
                            A + (long)(m0 + r) * lda + kb + c16);
        }
        // ---- stage B tile 32x64, transposed into BsT[n][k] ----
#pragma unroll
        for (int i = 0; i < 2; ++i) {
            int c  = tid + 256 * i;       // 0..511 chunks
            int kr = c >> 4;              // 0..31
            int n4 = (c & 15) * 4;
            uint2 d = *(const uint2*)(B + (long)(kb + kr) * ldb + n0 + n4);
            Bs[(n4 + 0) * GLD + kr] = (unsigned short)(d.x & 0xffffu);
            Bs[(n4 + 1) * GLD + kr] = (unsigned short)(d.x >> 16);
            Bs[(n4 + 2) * GLD + kr] = (unsigned short)(d.y & 0xffffu);
            Bs[(n4 + 3) * GLD + kr] = (unsigned short)(d.y >> 16);
        }
        if (kb + GBK < K) {  // hint next tiles into cache (global_prefetch_b8)
            __builtin_prefetch(A + (long)(m0 + (tid >> 1)) * lda + kb + GBK, 0, 0);
            __builtin_prefetch(B + (long)(kb + GBK + (tid & 31)) * ldb + n0, 0, 0);
        }
        wait_async0();
        __syncthreads();

        v16bf a0 = frag_a_lds(As, wm0 + l16, 0, half, GLD);
        v16bf a1 = frag_a_lds(As, wm0 + 16 + l16, 0, half, GLD);
        v16bf b0 = frag_b_ldsT(Bs, wn0 + l16, half, GLD);
        v16bf b1 = frag_b_ldsT(Bs, wn0 + 16 + l16, half, GLD);
        acc[0][0] = wmma_bf16(a0, b0, acc[0][0]);
        acc[0][1] = wmma_bf16(a0, b1, acc[0][1]);
        acc[1][0] = wmma_bf16(a1, b0, acc[1][0]);
        acc[1][1] = wmma_bf16(a1, b1, acc[1][1]);
        __syncthreads();
    }

    // ---- epilogue: C layout lanes0-15 M=g, lanes16-31 M=8+g, N=l16 ----
#pragma unroll
    for (int i = 0; i < 2; ++i) {
#pragma unroll
        for (int j = 0; j < 2; ++j) {
            const int col = n0 + wn0 + j * 16 + l16;
#pragma unroll
            for (int g = 0; g < 8; ++g) {
                const int row = m0 + wm0 + i * 16 + half * 8 + g;
                float v = acc[i][j][g];
                if constexpr (EPI >= 1) v += bias[col];
                if constexpr (EPI == 1)
                    v = 0.5f * v * (1.0f + erff(v * 0.70710678118654752f));
                if constexpr (EPI == 2) v += resid[(long)row * ldc + col];
                const long off = cbase + (long)row * ldc + col;
                if constexpr (EPI == 2) outF[off] = v;
                else                    outH[off] = f2bf(v);
            }
        }
    }
}

// ---------------------------------------------------------------------------
// LayerNorm (fp32 in) -> bf16 out.  One block per row, D = 1024.
// ---------------------------------------------------------------------------
__global__ __launch_bounds__(256)
void layernorm_bf16_k(const float* __restrict__ x, const float* __restrict__ gamma,
                      const float* __restrict__ beta, unsigned short* __restrict__ out,
                      int D) {
    const int row = blockIdx.x, tid = threadIdx.x;
    const float* xr = x + (long)row * D;
    float v[4], s = 0.f, ss = 0.f;
#pragma unroll
    for (int i = 0; i < 4; ++i) {
        float f = xr[tid + 256 * i];
        v[i] = f; s += f; ss += f * f;
    }
    for (int o = 16; o > 0; o >>= 1) { s += __shfl_xor(s, o, 32); ss += __shfl_xor(ss, o, 32); }
    __shared__ float rs[8], rss[8];
    const int w = tid >> 5, lane = tid & 31;
    if (lane == 0) { rs[w] = s; rss[w] = ss; }
    __syncthreads();
    if (w == 0) {
        float a = lane < 8 ? rs[lane] : 0.f;
        float b = lane < 8 ? rss[lane] : 0.f;
        for (int o = 4; o > 0; o >>= 1) { a += __shfl_xor(a, o, 32); b += __shfl_xor(b, o, 32); }
        if (lane == 0) { rs[0] = a; rss[0] = b; }
    }
    __syncthreads();
    const float mean = rs[0] / (float)D;
    const float var  = rss[0] / (float)D - mean * mean;
    const float rstd = rsqrtf(var + 1e-5f);
#pragma unroll
    for (int i = 0; i < 4; ++i) {
        int c = tid + 256 * i;
        out[(long)row * D + c] = f2bf((v[i] - mean) * rstd * gamma[c] + beta[c]);
    }
}

// ---------------------------------------------------------------------------
// fp32 -> bf16 elementwise
// ---------------------------------------------------------------------------
__global__ __launch_bounds__(256)
void cvt_bf16_k(const float* __restrict__ in, unsigned short* __restrict__ out, long n) {
    long i = (long)blockIdx.x * 256 + threadIdx.x;
    if (i < n) out[i] = f2bf(in[i]);
}

// ---------------------------------------------------------------------------
// Fold per-head 64x64 hproj into out-proj:  Wc[h*64+d, e] = sum_e' whp[h,d,e'] * wo[h*64+e', e]
// ---------------------------------------------------------------------------
__global__ __launch_bounds__(256)
void combine_w_k(const float* __restrict__ wh, const float* __restrict__ wo,
                 unsigned short* __restrict__ Wc) {
    const int idx = blockIdx.x * 256 + threadIdx.x;  // 1024*1024 total
    const int row = idx >> 10, e = idx & 1023;
    const int h = row >> 6, d = row & 63;
    const float* whp = wh + (long)(h * 64 + d) * 64;
    const float* wop = wo + (long)(h * 64) * 1024 + e;
    float s = 0.f;
#pragma unroll 8
    for (int t = 0; t < 64; ++t) s += whp[t] * wop[(long)t * 1024];
    Wc[idx] = f2bf(s);
}

__global__ __launch_bounds__(256)
void combine_b_k(const float* __restrict__ bh, const float* __restrict__ wo,
                 const float* __restrict__ bo, float* __restrict__ bc) {
    const int e = blockIdx.x * 256 + threadIdx.x;
    if (e >= 1024) return;
    float s = bo[e];
    for (int j = 0; j < 1024; ++j) s += bh[j] * wo[(long)j * 1024 + e];
    bc[e] = s;
}

// ---------------------------------------------------------------------------
// Attention: one block = (b, h, 16 query rows). 8 waves / 256 threads.
// qkv layout: [H][B*S][192] bf16 (q|k|v in last dim). Scores live in LDS fp32.
// Output: o[b*S+s, h*64+d] bf16 (concat-head layout feeding W_comb GEMM).
// ---------------------------------------------------------------------------
__global__ __launch_bounds__(256)
void attn_kernel(const unsigned short* __restrict__ qkv,
                 unsigned short* __restrict__ o, int S, int B) {
    extern __shared__ char smem_raw[];
    float*          sc    = (float*)smem_raw;                 // [16][S]
    unsigned short* qs    = (unsigned short*)(sc + 16 * S);   // [16][72]
    float*          opart = (float*)(qs + 16 * 72);           // [4][16][16]
    float*          rinv  = opart + 4 * 16 * 16;              // [16]

    const int tid = threadIdx.x, lane = tid & 31, w = tid >> 5;
    const int half = lane >> 4, l16 = lane & 15;
    const int q0 = blockIdx.x * 16, h = blockIdx.y, b = blockIdx.z;
    const int T = B * S;

    const unsigned short* Q  = qkv + ((long)h * T + (long)b * S) * 192;
    const unsigned short* Kp = Q + 64;
    const unsigned short* Vp = Q + 128;
    const float scale = 0.125f;  // 1/sqrt(64)

    // ---- load 16x32 Q tile (head dims 0..31) into LDS via async copy (64 x 16B chunks) ----
    if (tid < 64) {
        const int r = tid >> 2, c16 = (tid & 3) * 8;
        async_copy_b128(qs + r * 72 + c16, Q + (long)(q0 + r) * 192 + c16);
    }
    wait_async0();
    __syncthreads();

    // ---- phase 1: S = Q K^T * scale ; waves stripe over key tiles of 16 ----
    const v8f vzero = {0.f, 0.f, 0.f, 0.f, 0.f, 0.f, 0.f, 0.f};
    for (int kt = w; kt < (S >> 4); kt += 8) {
        v8f c = vzero;
#pragma unroll
        for (int ks = 0; ks < 2; ++ks) {
            const int k0 = ks * 32;
            v16bf a = frag_a_lds(qs, l16, k0, half, 72);
            // B frag = K^T: column N = key row; contiguous 16 bf16 along hd.
            const unsigned short* kr = Kp + (long)(kt * 16 + l16) * 192 + k0 + half * 16;
            u16x8 lo = *(const u16x8*)kr;
            u16x8 hi = *(const u16x8*)(kr + 8);
            v16u bu;
#pragma unroll
            for (int i = 0; i < 8; ++i) { bu[i] = lo[i]; bu[i + 8] = hi[i]; }
            c = wmma_bf16(a, __builtin_bit_cast(v16bf, bu), c);
        }
#pragma unroll
        for (int g = 0; g < 8; ++g)
            sc[(half * 8 + g) * S + kt * 16 + l16] = c[g] * scale;
    }
    __syncthreads();

    // ---- phase 2: softmax over S per row; wave handles rows w and w+8 ----
#pragma unroll
    for (int rr = 0; rr < 2; ++rr) {
        const int r = w + rr * 8;
        float* srow = sc + (long)r * S;
        float mx = -3.0e38f;
        for (int j = lane; j < S; j += 32) mx = fmaxf(mx, srow[j]);
        for (int o2 = 16; o2 > 0; o2 >>= 1) mx = fmaxf(mx, __shfl_xor(mx, o2, 32));
        float sum = 0.f;
        for (int j = lane; j < S; j += 32) {
            float p = __expf(srow[j] - mx);
            srow[j] = p; sum += p;
        }
        for (int o2 = 16; o2 > 0; o2 >>= 1) sum += __shfl_xor(sum, o2, 32);
        if (lane == 0) rinv[r] = 1.0f / sum;
    }
    __syncthreads();

    // ---- phase 3: O = P V (split-K over 2 waves per 16-wide N tile) ----
    const int nt = w & 3;        // N tile (output dims 0..63 in steps of 16)
    const int kh = w >> 2;       // key half
    v8f c = vzero;
    for (int k0 = kh * (S >> 1); k0 < (kh + 1) * (S >> 1); k0 += 32) {
        v16u au;
        const float* pr = sc + (long)l16 * S + k0;
#pragma unroll
        for (int i = 0; i < 8; ++i) au[i] = f2bf(pr[half * 16 + i]);
#pragma unroll
        for (int i = 0; i < 8; ++i) au[i + 8] = f2bf(pr[half * 16 + 8 + i]);
        v16u bu;
        const unsigned short* vcol = Vp + (long)(k0 + half * 16) * 192 + nt * 16 + l16;
#pragma unroll
        for (int i = 0; i < 16; ++i) bu[i] = vcol[(long)i * 192];
        c = wmma_bf16(__builtin_bit_cast(v16bf, au), __builtin_bit_cast(v16bf, bu), c);
    }
    if (kh == 1) {
#pragma unroll
        for (int g = 0; g < 8; ++g)
            opart[(nt * 16 + half * 8 + g) * 16 + l16] = c[g];
    }
    __syncthreads();
    if (kh == 0) {
#pragma unroll
        for (int g = 0; g < 8; ++g) {
            const int m = half * 8 + g;
            const float v = (c[g] + opart[(nt * 16 + m) * 16 + l16]) * rinv[m];
            const int col = h * 64 + nt * 16 + l16;
            o[((long)b * S + q0 + m) * 1024 + col] = f2bf(v);
        }
    }
}

// ---------------------------------------------------------------------------
// Host driver
// ---------------------------------------------------------------------------
extern "C" void kernel_launch(void* const* d_in, const int* in_sizes, int n_in,
                              void* d_out, int out_size, void* d_ws, size_t ws_size,
                              hipStream_t stream) {
    const float* x       = (const float*)d_in[0];
    const float* w_qkv   = (const float*)d_in[1];
    const float* w_hproj = (const float*)d_in[2];
    const float* b_hproj = (const float*)d_in[3];
    const float* w_out   = (const float*)d_in[4];
    const float* b_out   = (const float*)d_in[5];
    const float* w_fc1   = (const float*)d_in[6];
    const float* b_fc1   = (const float*)d_in[7];
    const float* w_fc2   = (const float*)d_in[8];
    const float* b_fc2   = (const float*)d_in[9];
    const float* g1      = (const float*)d_in[10];
    const float* be1     = (const float*)d_in[11];
    const float* g2      = (const float*)d_in[12];
    const float* be2     = (const float*)d_in[13];

    const int D = 1024, S = 2048, H = 16, F = 4096;
    const int T = in_sizes[0] / D;   // B*S = 4096
    const int B = T / S;

    // workspace carve
    char* ws = (char*)d_ws;
    size_t off = 0;
    auto carve = [&](size_t bytes) -> char* {
        char* p = ws + off;
        off = (off + bytes + 255) & ~((size_t)255);
        return p;
    };
    unsigned short* xn   = (unsigned short*)carve((size_t)T * D * 2);
    unsigned short* wqh  = (unsigned short*)carve((size_t)H * D * 192 * 2);
    unsigned short* qkv  = (unsigned short*)carve((size_t)H * T * 192 * 2);
    unsigned short* oc   = (unsigned short*)carve((size_t)T * D * 2);
    unsigned short* Wc   = (unsigned short*)carve((size_t)D * D * 2);
    float*          bc   = (float*)carve((size_t)D * 4);
    float*          x1   = (float*)carve((size_t)T * D * 4);
    unsigned short* xn2  = (unsigned short*)carve((size_t)T * D * 2);
    unsigned short* wf1  = (unsigned short*)carve((size_t)D * F * 2);
    unsigned short* wf2  = (unsigned short*)carve((size_t)F * D * 2);
    unsigned short* hbuf = (unsigned short*)carve((size_t)T * F * 2);

    // LN1 -> bf16
    layernorm_bf16_k<<<T, 256, 0, stream>>>(x, g1, be1, xn, D);

    // weight conversions + hproj/out-proj fusion
    const long nq = (long)H * D * 192;
    cvt_bf16_k<<<(int)((nq + 255) / 256), 256, 0, stream>>>(w_qkv, wqh, nq);
    const long nf = (long)D * F;
    cvt_bf16_k<<<(int)((nf + 255) / 256), 256, 0, stream>>>(w_fc1, wf1, nf);
    cvt_bf16_k<<<(int)((nf + 255) / 256), 256, 0, stream>>>(w_fc2, wf2, nf);
    combine_w_k<<<(D * D) / 256, 256, 0, stream>>>(w_hproj, w_out, Wc);
    combine_b_k<<<(D + 255) / 256, 256, 0, stream>>>(b_hproj, w_out, b_out, bc);

    // QKV: batched over heads, M=T, N=192, K=D
    {
        dim3 g(192 / GBN, T / GBM, H);
        gemm_bf16_wmma<0><<<g, 256, 0, stream>>>(
            xn, wqh, nullptr, nullptr, nullptr, qkv,
            T, 192, D, D, 192, 192,
            0L, (long)D * 192, (long)T * 192);
    }

    // attention (scores resident in LDS)
    {
        const size_t smem = (size_t)16 * S * 4 + 16 * 72 * 2 + 4 * 16 * 16 * 4 + 64;
        hipFuncSetAttribute(reinterpret_cast<const void*>(attn_kernel),
                            hipFuncAttributeMaxDynamicSharedMemorySize, (int)smem);
        dim3 g(S / 16, H, B);
        attn_kernel<<<g, 256, smem, stream>>>(qkv, oc, S, B);
    }

    // fused (hproj -> concat -> out proj) + bias + residual -> x1 fp32
    {
        dim3 g(D / GBN, T / GBM, 1);
        gemm_bf16_wmma<2><<<g, 256, 0, stream>>>(
            oc, Wc, bc, x, x1, nullptr,
            T, D, D, D, D, D, 0L, 0L, 0L);
    }

    // LN2 -> bf16
    layernorm_bf16_k<<<T, 256, 0, stream>>>(x1, g2, be2, xn2, D);

    // fc1 + bias + exact GELU -> bf16
    {
        dim3 g(F / GBN, T / GBM, 1);
        gemm_bf16_wmma<1><<<g, 256, 0, stream>>>(
            xn2, wf1, b_fc1, nullptr, nullptr, hbuf,
            T, F, D, D, F, F, 0L, 0L, 0L);
    }

    // fc2 + bias + residual -> d_out fp32
    {
        dim3 g(D / GBN, T / GBM, 1);
        gemm_bf16_wmma<2><<<g, 256, 0, stream>>>(
            hbuf, wf2, b_fc2, x1, (float*)d_out, nullptr,
            T, D, F, F, D, D, 0L, 0L, 0L);
    }
}
